// Attention_65532611003000
// MI455X (gfx1250) — matrix operand order repaired and checked by reading, not compiled, not run on hardware
//
#include <hip/hip_runtime.h>
#include <hip/hip_bf16.h>

// ---------------------------------------------------------------------------
// Efficient (linear) attention, MI455X / gfx1250, wave32, bf16 WMMA path.
// B=8, N=4096, C=K=V=512, H=8, hk=hv=64.
// Round 3: vectorized (uint2) staging of the context/attended tiles,
// native packed f32->bf16 conversion when the builtin is available.
// ---------------------------------------------------------------------------

#define BB   8
#define NN   4096
#define CC   512
#define HH   8
#define HD   64          // head dim (hk == hv)
#define MROWS (BB*NN)    // 32768 token rows

typedef __attribute__((ext_vector_type(16))) __bf16 v16bf;
typedef __attribute__((ext_vector_type(2)))  __bf16 v2bf;
typedef __attribute__((ext_vector_type(8)))  float  v8f;

// ---- bf16 helpers (round-to-nearest-even) ---------------------------------
__device__ __forceinline__ unsigned short f32_to_bf16(float f) {
    unsigned int u = __builtin_bit_cast(unsigned int, f);
    unsigned int r = u + 0x7FFFu + ((u >> 16) & 1u);
    return (unsigned short)(r >> 16);
}
__device__ __forceinline__ float bf16_to_f32(unsigned short h) {
    unsigned int u = ((unsigned int)h) << 16;
    return __builtin_bit_cast(float, u);
}
__device__ __forceinline__ __bf16 us_to_bf(unsigned short h) {
    return __builtin_bit_cast(__bf16, h);
}
__device__ __forceinline__ unsigned int pack2_bf16(float lo, float hi) {
#if defined(__has_builtin)
#if __has_builtin(__builtin_amdgcn_cvt_pk_bf16_f32)
    v2bf p = __builtin_amdgcn_cvt_pk_bf16_f32(lo, hi);
    return __builtin_bit_cast(unsigned int, p);
#else
    return (unsigned int)f32_to_bf16(lo) | ((unsigned int)f32_to_bf16(hi) << 16);
#endif
#else
    return (unsigned int)f32_to_bf16(lo) | ((unsigned int)f32_to_bf16(hi) << 16);
#endif
}

// ---- WMMA wrapper ----------------------------------------------------------
__device__ __forceinline__ v8f wmma_bf16(v16bf a, v16bf b, v8f c) {
    // D = A(16x32 bf16) * B(32x16 bf16) + C(16x16 f32)
    return __builtin_amdgcn_wmma_f32_16x16x32_bf16(
        false, a, false, b, (short)0, c, false, false);
}

// ---- fragment loaders (layouts per CDNA5 ISA 7.12.2) -----------------------
// A 16x32 bf16 from row-major LDS [row][k]:
//   lane m = lane&15, halves t: k = g*8 + (t&7) + ((t&8)?16:0)
//   -> two contiguous 8-half (16B) runs per lane.
__device__ __forceinline__ v16bf load_A_frag(const unsigned short* lds,
                                             int row0, int k0, int stride, int lane) {
    const int m = lane & 15, g = lane >> 4;
    const unsigned short* p = lds + (row0 + m) * stride + k0 + g * 8;
    v16bf a;
#pragma unroll
    for (int t = 0; t < 16; ++t) {
        int k = (t & 7) + ((t & 8) ? 16 : 0);
        a[t] = us_to_bf(p[k]);
    }
    return a;
}
// B 32x16 bf16 from COLUMN-major LDS [col][k]:
//   lane n = lane&15, halves t: k = g*8 + (t&7) + ((t&8)?16:0)
//   -> two contiguous 8-half (16B) runs per lane.
__device__ __forceinline__ v16bf load_Bt_frag(const unsigned short* lds,
                                              int col0, int k0, int stride, int lane) {
    const int n = lane & 15, g = lane >> 4;
    const unsigned short* p = lds + (col0 + n) * stride + k0 + g * 8;
    v16bf b;
#pragma unroll
    for (int t = 0; t < 16; ++t) b[t] = us_to_bf(p[(t & 7) + ((t & 8) ? 16 : 0)]);
    return b;
}

// ---------------------------------------------------------------------------
// Kernel 0: fp32 -> bf16 weight conversion WITH transpose.
// src: [k][c] row-major (C x K);  dst: [c][k] (output-column major).
// ---------------------------------------------------------------------------
__global__ __launch_bounds__(256) void cvt_bf16_t_kernel(const float* __restrict__ src,
                                                         unsigned short* __restrict__ dst) {
    int i = blockIdx.x * 256 + threadIdx.x;     // i over C*C
    int c = i & (CC - 1), k = i >> 9;           // coalesced read along c
    dst[(long)c * CC + k] = f32_to_bf16(src[(long)k * CC + c]);
}

// ---------------------------------------------------------------------------
// Kernel 1: projection GEMM.  act(32768 x 512) @ W(512 x 512) + bias -> bf16
// mode 0: act = X+Y (keys), 1: act = Y (queries), 2: act = X (values)
// Block 256 thr (8 waves, 2x4), tile 128 x 256, K-step 32, double-buffered.
// ---------------------------------------------------------------------------
__global__ __launch_bounds__(256) void proj_gemm_kernel(
    const float* __restrict__ X, const float* __restrict__ Y,
    const unsigned short* __restrict__ WbT,   // [col][k] bf16
    const float* __restrict__ bias,
    unsigned short* __restrict__ Out, int mode)
{
    __shared__ unsigned short Alds[2][128 * 36];   // [row][k], stride 36
    __shared__ unsigned short Blds[2][256 * 36];   // [col][k], stride 36

    const int tid  = threadIdx.x;
    const int lane = tid & 31, wid = tid >> 5;
    const int wm = wid >> 2, wn = wid & 3;         // wave grid 2 x 4
    const int rowBase = blockIdx.y * 128;
    const int colBase = blockIdx.x * 256;

    v8f acc[4][4];
#pragma unroll
    for (int i = 0; i < 4; ++i)
#pragma unroll
        for (int j = 0; j < 4; ++j) acc[i][j] = v8f{};

    auto stage = [&](int buf, int kk) {
        // A: 128 rows x 32 k, float4 loads -> packed bf16 pairs -> b64 LDS stores
        for (int e = tid; e < 128 * 8; e += 256) {
            int r = e >> 3, c4 = (e & 7) * 4;
            long gi = (long)(rowBase + r) * CC + kk + c4;
            float4 v;
            if (mode == 0) {
                float4 a = *(const float4*)(X + gi);
                float4 b = *(const float4*)(Y + gi);
                v.x = a.x + b.x; v.y = a.y + b.y; v.z = a.z + b.z; v.w = a.w + b.w;
            } else if (mode == 1) {
                v = *(const float4*)(Y + gi);
            } else {
                v = *(const float4*)(X + gi);
            }
            uint2 p;
            p.x = pack2_bf16(v.x, v.y);
            p.y = pack2_bf16(v.z, v.w);
            *(uint2*)(&Alds[buf][r * 36 + c4]) = p;
        }
        // B: 256 cols x 32 k, contiguous bf16 in transposed global layout
        for (int e = tid; e < 256 * 8; e += 256) {
            int c = e >> 3, k4 = (e & 7) * 4;
            uint2 w = *(const uint2*)(WbT + (long)(colBase + c) * CC + kk + k4);
            *(uint2*)(&Blds[buf][c * 36 + k4]) = w;
        }
    };

    stage(0, 0);
    for (int kk = 0; kk < CC; kk += 32) {
        const int buf = (kk >> 5) & 1;
        __syncthreads();
        if (kk + 32 < CC) {
            long pgi = (long)(rowBase + (tid & 127)) * CC + kk + 32;
            __builtin_prefetch(X + pgi, 0, 1);
            stage(buf ^ 1, kk + 32);
        }
        v16bf afr[4], bfr[4];
#pragma unroll
        for (int i = 0; i < 4; ++i)
            afr[i] = load_A_frag(&Alds[buf][0], wm * 64 + i * 16, 0, 36, lane);
#pragma unroll
        for (int j = 0; j < 4; ++j)
            bfr[j] = load_Bt_frag(&Blds[buf][0], wn * 64 + j * 16, 0, 36, lane);
#pragma unroll
        for (int i = 0; i < 4; ++i)
#pragma unroll
            for (int j = 0; j < 4; ++j)
                acc[i][j] = wmma_bf16(afr[i], bfr[j], acc[i][j]);
    }

    // epilogue: + bias, store bf16
    const int g = lane >> 4, ln = lane & 15;
#pragma unroll
    for (int i = 0; i < 4; ++i)
#pragma unroll
        for (int j = 0; j < 4; ++j) {
            int col = colBase + wn * 64 + j * 16 + ln;
            float bcol = bias[col];
#pragma unroll
            for (int r = 0; r < 8; ++r) {
                int row = rowBase + wm * 64 + i * 16 + g * 8 + r;
                Out[(long)row * CC + col] = f32_to_bf16(acc[i][j][r] + bcol);
            }
        }
}

// ---------------------------------------------------------------------------
// Kernel 2: key softmax over tokens N (axis=1), in place on bf16 keys.
// One thread per (b,k) column; grid (B, 2), block 256 (coalesced over k).
// ---------------------------------------------------------------------------
__global__ __launch_bounds__(256) void key_softmax_kernel(unsigned short* __restrict__ keys)
{
    int b = blockIdx.x;
    int k = blockIdx.y * 256 + threadIdx.x;
    unsigned short* col = keys + (long)b * NN * CC + k;
    float mx = -1e30f;
    for (int n = 0; n < NN; ++n)
        mx = fmaxf(mx, bf16_to_f32(col[(long)n * CC]));
    float s = 0.f;
    for (int n = 0; n < NN; ++n)
        s += __expf(bf16_to_f32(col[(long)n * CC]) - mx);
    float inv = 1.0f / s;
    for (int n = 0; n < NN; ++n) {
        float v = __expf(bf16_to_f32(col[(long)n * CC]) - mx) * inv;
        col[(long)n * CC] = f32_to_bf16(v);
    }
}

// ---------------------------------------------------------------------------
// Kernel 3: query softmax over the 64 head channels, in place on bf16 q.
// One wave per (b,n,h) group (64 contiguous bf16), 2 channels per lane.
// ---------------------------------------------------------------------------
__global__ __launch_bounds__(256) void query_softmax_kernel(unsigned short* __restrict__ q)
{
    const int wid = threadIdx.x >> 5, lane = threadIdx.x & 31;
    long grp = (long)blockIdx.x * 8 + wid;          // B*N*H = 262144 groups
    unsigned short* p = q + grp * HD;
    float v0 = bf16_to_f32(p[2 * lane]);
    float v1 = bf16_to_f32(p[2 * lane + 1]);
    float mx = fmaxf(v0, v1);
#pragma unroll
    for (int off = 16; off > 0; off >>= 1) mx = fmaxf(mx, __shfl_xor(mx, off, 32));
    float e0 = __expf(v0 - mx), e1 = __expf(v1 - mx);
    float s = e0 + e1;
#pragma unroll
    for (int off = 16; off > 0; off >>= 1) s += __shfl_xor(s, off, 32);
    float inv = 1.0f / s;
    p[2 * lane]     = f32_to_bf16(e0 * inv);
    p[2 * lane + 1] = f32_to_bf16(e1 * inv);
}

// ---------------------------------------------------------------------------
// Kernel 4: context[b,h] = keys[b,:,h]^T (64 x 4096) @ values[b,:,h] (4096 x 64)
// One block (128 thr = 4 waves) per (b,h); wave w owns k-rows [16w,16w+16).
// Tiles staged channel-major in LDS (contiguous fragment reads);
// global reads vectorized uint2 (4 bf16 per access); double-buffered over N.
// ---------------------------------------------------------------------------
__global__ __launch_bounds__(128) void context_kernel(
    const unsigned short* __restrict__ keys, const unsigned short* __restrict__ vals,
    unsigned short* __restrict__ ctx)
{
    const int bh = blockIdx.x;                 // 64 blocks
    const int b = bh >> 3, h = bh & 7;
    __shared__ unsigned short Kt[2][64 * 40];  // [k_channel][n_local], stride 40
    __shared__ unsigned short Vt[2][64 * 40];  // [v_channel][n_local]
    const int tid = threadIdx.x, lane = tid & 31, w = tid >> 5;
    const int m = lane & 15, g = lane >> 4;

    v8f acc[4];
#pragma unroll
    for (int j = 0; j < 4; ++j) acc[j] = v8f{};

    const long base = (long)b * NN * CC + h * HD;
    auto stageKV = [&](int buf, int n0) {
        // 32 rows x 16 uint2-chunks (4 channels each); coalesced over chunks
        for (int e = tid; e < 32 * 16; e += 128) {
            int r = e >> 4, c4 = (e & 15) * 4;     // r = n_local, c4 = channel base
            long gi = base + (long)(n0 + r) * CC + c4;
            uint2 kv = *(const uint2*)(keys + gi);
            uint2 vv = *(const uint2*)(vals + gi);
            Kt[buf][(c4 + 0) * 40 + r] = (unsigned short)(kv.x & 0xFFFF);
            Kt[buf][(c4 + 1) * 40 + r] = (unsigned short)(kv.x >> 16);
            Kt[buf][(c4 + 2) * 40 + r] = (unsigned short)(kv.y & 0xFFFF);
            Kt[buf][(c4 + 3) * 40 + r] = (unsigned short)(kv.y >> 16);
            Vt[buf][(c4 + 0) * 40 + r] = (unsigned short)(vv.x & 0xFFFF);
            Vt[buf][(c4 + 1) * 40 + r] = (unsigned short)(vv.x >> 16);
            Vt[buf][(c4 + 2) * 40 + r] = (unsigned short)(vv.y & 0xFFFF);
            Vt[buf][(c4 + 3) * 40 + r] = (unsigned short)(vv.y >> 16);
        }
    };

    stageKV(0, 0);
    for (int n0 = 0; n0 < NN; n0 += 32) {
        const int buf = (n0 >> 5) & 1;
        __syncthreads();
        if (n0 + 32 < NN) stageKV(buf ^ 1, n0 + 32);

        // A = keys^T: rows = k channel (16w+m), reduction dim = n (contiguous).
        v16bf a = load_A_frag(&Kt[buf][0], 16 * w, 0, 40, lane);
#pragma unroll
        for (int j = 0; j < 4; ++j) {
            // B = values: cols = v channel, reduction dim = n (contiguous).
            v16bf bf = load_Bt_frag(&Vt[buf][0], j * 16, 0, 40, lane);
            acc[j] = wmma_bf16(a, bf, acc[j]);
        }
    }

    // store context head as bf16 (64 x 64)
#pragma unroll
    for (int j = 0; j < 4; ++j)
#pragma unroll
        for (int r = 0; r < 8; ++r) {
            int krow = 16 * w + g * 8 + r;
            int vcol = j * 16 + m;
            ctx[((long)bh * HD + krow) * HD + vcol] = f32_to_bf16(acc[j][r]);
        }
}

// ---------------------------------------------------------------------------
// Kernel 5: attended[b,n,h,:] = q[b,n,h,:] @ context[b,h]  (block-diagonal GEMM)
// Block 256 thr (8 waves) per 128-token tile; wave w owns rows [16w,16w+16).
// ctx staged [v][k] in LDS (vectorized global reads); q staged uint2.
// ---------------------------------------------------------------------------
__global__ __launch_bounds__(256) void attended_kernel(
    const unsigned short* __restrict__ q, const unsigned short* __restrict__ ctx,
    unsigned short* __restrict__ att)
{
    const int tid = threadIdx.x, lane = tid & 31, w = tid >> 5;
    const int rowBase = blockIdx.x * 128;
    const int b = rowBase >> 12;               // 4096 tokens per batch, 128 | 4096
    const int g = lane >> 4, ln = lane & 15;

    __shared__ unsigned short CtxT[64 * 68];   // [v][k], stride 68
    __shared__ unsigned short Qlds[128 * 68];  // [row][k], stride 68

    for (int h = 0; h < HH; ++h) {
        // stage ctx head transposed: vector reads along v, scatter to [v][k]
        for (int e = tid; e < 64 * 16; e += 256) {
            int r = e >> 4, c4 = (e & 15) * 4;     // r = k row, c4 = v col base
            uint2 v = *(const uint2*)(ctx + (((long)b * HH + h) * HD + r) * HD + c4);
            CtxT[(c4 + 0) * 68 + r] = (unsigned short)(v.x & 0xFFFF);
            CtxT[(c4 + 1) * 68 + r] = (unsigned short)(v.x >> 16);
            CtxT[(c4 + 2) * 68 + r] = (unsigned short)(v.y & 0xFFFF);
            CtxT[(c4 + 3) * 68 + r] = (unsigned short)(v.y >> 16);
        }
        for (int e = tid; e < 128 * 16; e += 256) {
            int r = e >> 4, c4 = (e & 15) * 4;
            uint2 v = *(const uint2*)(q + (long)(rowBase + r) * CC + h * HD + c4);
            *(uint2*)(&Qlds[r * 68 + c4]) = v;
        }
        __syncthreads();

        v8f acc[4];
#pragma unroll
        for (int j = 0; j < 4; ++j) acc[j] = v8f{};
#pragma unroll
        for (int kc = 0; kc < HD; kc += 32) {
            v16bf a = load_A_frag(Qlds, 16 * w, kc, 68, lane);
#pragma unroll
            for (int j = 0; j < 4; ++j) {
                v16bf bf = load_Bt_frag(CtxT, j * 16, kc, 68, lane);
                acc[j] = wmma_bf16(a, bf, acc[j]);
            }
        }
        __syncthreads();   // all waves done reading LDS before next head restage

#pragma unroll
        for (int j = 0; j < 4; ++j)
#pragma unroll
            for (int r = 0; r < 8; ++r) {
                int row = rowBase + 16 * w + g * 8 + r;
                int col = h * HD + j * 16 + ln;
                att[(long)row * CC + col] = f32_to_bf16(acc[j][r]);
            }
    }
}

// ---------------------------------------------------------------------------
// Kernel 6: reprojection GEMM. att(bf16) @ Wr(bf16,[col][k]) + br -> f32 out
// Same tiling as proj_gemm, bf16 activation input, double-buffered.
// ---------------------------------------------------------------------------
__global__ __launch_bounds__(256) void reproj_gemm_kernel(
    const unsigned short* __restrict__ Act, const unsigned short* __restrict__ WbT,
    const float* __restrict__ bias, float* __restrict__ Out)
{
    __shared__ unsigned short Alds[2][128 * 36];
    __shared__ unsigned short Blds[2][256 * 36];

    const int tid  = threadIdx.x;
    const int lane = tid & 31, wid = tid >> 5;
    const int wm = wid >> 2, wn = wid & 3;
    const int rowBase = blockIdx.y * 128;
    const int colBase = blockIdx.x * 256;

    v8f acc[4][4];
#pragma unroll
    for (int i = 0; i < 4; ++i)
#pragma unroll
        for (int j = 0; j < 4; ++j) acc[i][j] = v8f{};

    auto stage = [&](int buf, int kk) {
        for (int e = tid; e < 128 * 8; e += 256) {
            int r = e >> 3, c4 = (e & 7) * 4;
            uint2 v = *(const uint2*)(Act + (long)(rowBase + r) * CC + kk + c4);
            *(uint2*)(&Alds[buf][r * 36 + c4]) = v;
        }
        for (int e = tid; e < 256 * 8; e += 256) {
            int c = e >> 3, k4 = (e & 7) * 4;
            uint2 w = *(const uint2*)(WbT + (long)(colBase + c) * CC + kk + k4);
            *(uint2*)(&Blds[buf][c * 36 + k4]) = w;
        }
    };

    stage(0, 0);
    for (int kk = 0; kk < CC; kk += 32) {
        const int buf = (kk >> 5) & 1;
        __syncthreads();
        if (kk + 32 < CC) {
            __builtin_prefetch(Act + (long)(rowBase + (tid & 127)) * CC + kk + 32, 0, 1);
            stage(buf ^ 1, kk + 32);
        }
        v16bf afr[4], bfr[4];
#pragma unroll
        for (int i = 0; i < 4; ++i)
            afr[i] = load_A_frag(&Alds[buf][0], wm * 64 + i * 16, 0, 36, lane);
#pragma unroll
        for (int j = 0; j < 4; ++j)
            bfr[j] = load_Bt_frag(&Blds[buf][0], wn * 64 + j * 16, 0, 36, lane);
#pragma unroll
        for (int i = 0; i < 4; ++i)
#pragma unroll
            for (int j = 0; j < 4; ++j)
                acc[i][j] = wmma_bf16(afr[i], bfr[j], acc[i][j]);
    }

    const int g = lane >> 4, ln = lane & 15;
#pragma unroll
    for (int i = 0; i < 4; ++i)
#pragma unroll
        for (int j = 0; j < 4; ++j) {
            int col = colBase + wn * 64 + j * 16 + ln;
            float bcol = bias[col];
#pragma unroll
            for (int r = 0; r < 8; ++r) {
                int row = rowBase + wm * 64 + i * 16 + g * 8 + r;
                Out[(long)row * CC + col] = acc[i][j][r] + bcol;
            }
        }
}

// ---------------------------------------------------------------------------
// Host-side orchestration
// ---------------------------------------------------------------------------
extern "C" void kernel_launch(void* const* d_in, const int* in_sizes, int n_in,
                              void* d_out, int out_size, void* d_ws, size_t ws_size,
                              hipStream_t stream) {
    (void)in_sizes; (void)n_in; (void)out_size; (void)ws_size;

    const float* X  = (const float*)d_in[0];   // input_
    const float* Y  = (const float*)d_in[1];   // y
    const float* Wk = (const float*)d_in[2];
    const float* bk = (const float*)d_in[3];
    const float* Wq = (const float*)d_in[4];
    const float* bq = (const float*)d_in[5];
    const float* Wv = (const float*)d_in[6];
    const float* bv = (const float*)d_in[7];
    const float* Wr = (const float*)d_in[8];
    const float* br = (const float*)d_in[9];
    float* Out = (float*)d_out;

    // workspace carve-up (256B aligned)
    size_t off = 0;
    auto carve = [&](size_t bytes) -> void* {
        void* p = (char*)d_ws + off;
        off += (bytes + 255) & ~(size_t)255;
        return p;
    };
    const size_t wbytes   = (size_t)CC * CC * 2;          // 512 KB each
    const size_t actbytes = (size_t)MROWS * CC * 2;       // 32 MB each
    unsigned short* WkT = (unsigned short*)carve(wbytes);
    unsigned short* WqT = (unsigned short*)carve(wbytes);
    unsigned short* WvT = (unsigned short*)carve(wbytes);
    unsigned short* WrT = (unsigned short*)carve(wbytes);
    unsigned short* Kb  = (unsigned short*)carve(actbytes);   // keys bf16
    unsigned short* Qb  = (unsigned short*)carve(actbytes);   // queries bf16
    unsigned short* Vb  = (unsigned short*)carve(actbytes);   // values bf16
    unsigned short* Ctx = (unsigned short*)carve((size_t)BB * HH * HD * HD * 2);
    unsigned short* Att = (unsigned short*)carve(actbytes);   // attended bf16

    dim3 cvtGrid((CC * CC) / 256);
    cvt_bf16_t_kernel<<<cvtGrid, 256, 0, stream>>>(Wk, WkT);
    cvt_bf16_t_kernel<<<cvtGrid, 256, 0, stream>>>(Wq, WqT);
    cvt_bf16_t_kernel<<<cvtGrid, 256, 0, stream>>>(Wv, WvT);
    cvt_bf16_t_kernel<<<cvtGrid, 256, 0, stream>>>(Wr, WrT);

    dim3 gemmGrid(CC / 256, MROWS / 128);   // (2, 256)
    proj_gemm_kernel<<<gemmGrid, 256, 0, stream>>>(X, Y, WkT, bk, Kb, 0);  // keys: x+y
    proj_gemm_kernel<<<gemmGrid, 256, 0, stream>>>(X, Y, WqT, bq, Qb, 1);  // queries: y
    proj_gemm_kernel<<<gemmGrid, 256, 0, stream>>>(X, Y, WvT, bv, Vb, 2);  // values: x

    key_softmax_kernel<<<dim3(BB, CC / 256), 256, 0, stream>>>(Kb);
    query_softmax_kernel<<<dim3((BB * NN * HH) / 8), 256, 0, stream>>>(Qb);

    context_kernel<<<dim3(BB * HH), 128, 0, stream>>>(Kb, Vb, Ctx);
    attended_kernel<<<dim3(MROWS / 128), 256, 0, stream>>>(Qb, Ctx, Att);

    reproj_gemm_kernel<<<gemmGrid, 256, 0, stream>>>(Att, WrT, br, Out);
}
